// K_L_RAG_9440338116826
// MI455X (gfx1250) — matrix operand order repaired and checked by reading, not compiled, not run on hardware
//
#include <hip/hip_runtime.h>
#include <hip/hip_bf16.h>
#include <math.h>

typedef __attribute__((ext_vector_type(16))) __bf16 bf16x16;
typedef __attribute__((ext_vector_type(8)))  __bf16 bf16x8;
typedef __attribute__((ext_vector_type(4)))  __bf16 bf16x4;
typedef __attribute__((ext_vector_type(8)))  float  f32x8;

#define D_MODEL 2048
#define BROWS   8192
#define NTOK    2
#define KNOW    100

static __device__ __forceinline__ bf16x16 combine16(bf16x8 lo, bf16x8 hi) {
  return __builtin_shufflevector(lo, hi, 0,1,2,3,4,5,6,7,8,9,10,11,12,13,14,15);
}

#define WMMA_BF16(A_, B_, C_)                                                  \
  __builtin_amdgcn_wmma_f32_16x16x32_bf16(false, (A_), false, (B_),            \
                                          (short)0, (C_), false, false)

// ---------------------------------------------------------------------------
// fp32 -> bf16 conversion (vectorized x4)
// ---------------------------------------------------------------------------
__global__ __launch_bounds__(256) void cvt_f32_bf16(const float* __restrict__ s,
                                                    __bf16* __restrict__ d, int n4) {
  int i = blockIdx.x * 256 + threadIdx.x;
  if (i >= n4) return;
  float4 v = ((const float4*)s)[i];
  bf16x4 o;
  o[0] = (__bf16)v.x; o[1] = (__bf16)v.y; o[2] = (__bf16)v.z; o[3] = (__bf16)v.w;
  ((bf16x4*)d)[i] = o;
}

// ---------------------------------------------------------------------------
// C[M,N] = A[M,K] @ W[N,K]^T + bias   (bf16 in, f32 accumulate via WMMA)
// Per wave: 32(M) x 64(N) tile = 8 accumulators, software-pipelined K loop
// (next k-step's 6 fragments are in flight while the 8 WMMAs of the current
// step execute).  8 waves/block stacked in M -> 256 x 64 block tile.
// MODE 0: bf16 row-major store.  MODE 1: fp32 store with (B,2,d)->(2,B,d)
// permutation folded in.
// ---------------------------------------------------------------------------
template<int MODE>
__global__ __launch_bounds__(256) void gemm_nt_bf16(
    const __bf16* __restrict__ A,
    const __bf16* __restrict__ W,
    const float*  __restrict__ bias,
    void* __restrict__ Cout,
    int M, int N, int K)
{
  const int lane = threadIdx.x & 31;
  const int wave = threadIdx.x >> 5;
  const int hi   = lane >> 4;
  const int lx   = lane & 15;
  const int m0   = (blockIdx.x * 8 + wave) * 32;
  const int n0   = blockIdx.y * 64;

  // A fragment per ISA 16-bit A layout: lane<16 -> K {0..7,16..23}, lane>=16 -> {8..15,24..31}
  const __bf16* pa0 = A + (size_t)(m0 + lx) * K + (hi << 3);
  const __bf16* pa1 = pa0 + (size_t)16 * K;
  // B fragment: lane<16 -> K {0..7,16..23}, lane>=16 -> {8..15,24..31}
  const __bf16* pb  = W + (size_t)(n0 + lx) * K + (hi << 3);
  const size_t  bstride = (size_t)16 * K;

  f32x8 acc[8] = {};

  auto loadA = [&](const __bf16* p, int k) -> bf16x16 {
    return combine16(*(const bf16x8*)(p + k), *(const bf16x8*)(p + k + 16));
  };
  auto loadB = [&](int t, int k) -> bf16x16 {
    const __bf16* p = pb + (size_t)t * bstride + k;
    return combine16(*(const bf16x8*)p, *(const bf16x8*)(p + 16));
  };

  // pipeline prologue: fragments for k = 0
  bf16x16 a0 = loadA(pa0, 0), a1 = loadA(pa1, 0);
  bf16x16 b0 = loadB(0, 0), b1 = loadB(1, 0), b2 = loadB(2, 0), b3 = loadB(3, 0);

  for (int k = 32; k < K; k += 32) {
    // issue next step's loads before this step's math
    bf16x16 na0 = loadA(pa0, k), na1 = loadA(pa1, k);
    bf16x16 nb0 = loadB(0, k), nb1 = loadB(1, k);
    bf16x16 nb2 = loadB(2, k), nb3 = loadB(3, k);

    acc[0] = WMMA_BF16(a0, b0, acc[0]);
    acc[1] = WMMA_BF16(a0, b1, acc[1]);
    acc[2] = WMMA_BF16(a0, b2, acc[2]);
    acc[3] = WMMA_BF16(a0, b3, acc[3]);
    acc[4] = WMMA_BF16(a1, b0, acc[4]);
    acc[5] = WMMA_BF16(a1, b1, acc[5]);
    acc[6] = WMMA_BF16(a1, b2, acc[6]);
    acc[7] = WMMA_BF16(a1, b3, acc[7]);

    a0 = na0; a1 = na1;
    b0 = nb0; b1 = nb1; b2 = nb2; b3 = nb3;
  }
  // pipeline epilogue: last k-step
  acc[0] = WMMA_BF16(a0, b0, acc[0]);
  acc[1] = WMMA_BF16(a0, b1, acc[1]);
  acc[2] = WMMA_BF16(a0, b2, acc[2]);
  acc[3] = WMMA_BF16(a0, b3, acc[3]);
  acc[4] = WMMA_BF16(a1, b0, acc[4]);
  acc[5] = WMMA_BF16(a1, b1, acc[5]);
  acc[6] = WMMA_BF16(a1, b2, acc[6]);
  acc[7] = WMMA_BF16(a1, b3, acc[7]);

#pragma unroll
  for (int mt = 0; mt < 2; ++mt) {
#pragma unroll
    for (int t = 0; t < 4; ++t) {
      const f32x8 av   = acc[mt * 4 + t];
      const int   col  = n0 + t * 16 + lx;
      const float bvv  = bias[col];
      const int  rbase = m0 + mt * 16 + hi * 8;
      if (MODE == 0) {
        __bf16* C = (__bf16*)Cout;
#pragma unroll
        for (int r = 0; r < 8; ++r)
          C[(size_t)(rbase + r) * N + col] = (__bf16)(av[r] + bvv);
      } else {
        float* C = (float*)Cout;
        const int tok = col >> 11;            // col / D_MODEL
        const int dd  = col & (D_MODEL - 1);  // col % D_MODEL
#pragma unroll
        for (int r = 0; r < 8; ++r)
          C[((size_t)tok * BROWS + (rbase + r)) * D_MODEL + dd] = av[r] + bvv;
      }
    }
  }
}

// ---------------------------------------------------------------------------
// Fused: scores = q @ K_keys^T / sqrt(d); top-3; softmax; retrieved = w.K_emb
// One wave per row; q row staged in LDS; per-lane redundant top-3 scan.
// ---------------------------------------------------------------------------
__global__ __launch_bounds__(256) void retrieve_topk(
    const __bf16* __restrict__ q,      // B x D bf16
    const float*  __restrict__ Kkeys,  // KNOW x D fp32
    const float*  __restrict__ Kemb,   // KNOW x D fp32
    __bf16* __restrict__ out)          // B x D bf16
{
  __shared__ __bf16 sq[8][D_MODEL];
  __shared__ float  ssc[8][128];
  const int lane = threadIdx.x & 31;
  const int wave = threadIdx.x >> 5;
  const int row  = blockIdx.x * 8 + wave;

  {
    bf16x8*       dst = (bf16x8*)sq[wave];
    const bf16x8* src = (const bf16x8*)(q + (size_t)row * D_MODEL);
#pragma unroll
    for (int i = 0; i < (D_MODEL / 8) / 32; ++i)
      dst[lane + i * 32] = src[lane + i * 32];
  }
  for (int j = lane; j < 128; j += 32) ssc[wave][j] = -3.4e38f;
  __syncthreads();

  const float scale = 0.022097086912079608f;  // 1/sqrt(2048)
  for (int j = lane; j < KNOW; j += 32) {
    const float*  kk = Kkeys + (size_t)j * D_MODEL;
    const bf16x8* sv = (const bf16x8*)sq[wave];
    float acc = 0.f;
    for (int c = 0; c < D_MODEL / 8; ++c) {
      bf16x8 qa = sv[c];
      const float* kp = kk + c * 8;
#pragma unroll
      for (int e = 0; e < 8; ++e) acc += (float)qa[e] * kp[e];
    }
    ssc[wave][j] = acc * scale;
  }
  __syncthreads();

  float s0 = -3.4e38f, s1 = -3.4e38f, s2 = -3.4e38f;
  int   i0 = 0, i1 = 0, i2 = 0;
  for (int j = 0; j < KNOW; ++j) {
    float s = ssc[wave][j];
    if (s > s0)      { s2 = s1; i2 = i1; s1 = s0; i1 = i0; s0 = s; i0 = j; }
    else if (s > s1) { s2 = s1; i2 = i1; s1 = s;  i1 = j; }
    else if (s > s2) { s2 = s;  i2 = j; }
  }
  float e0 = 1.f, e1 = __expf(s1 - s0), e2 = __expf(s2 - s0);
  float inv = 1.f / (e0 + e1 + e2);
  float w0 = e0 * inv, w1 = e1 * inv, w2 = e2 * inv;

  const float* p0 = Kemb + (size_t)i0 * D_MODEL;
  const float* p1 = Kemb + (size_t)i1 * D_MODEL;
  const float* p2 = Kemb + (size_t)i2 * D_MODEL;
  __bf16* orow = out + (size_t)row * D_MODEL;
  for (int d = lane; d < D_MODEL; d += 32)
    orow[d] = (__bf16)(w0 * p0[d] + w1 * p1[d] + w2 * p2[d]);
}

// ---------------------------------------------------------------------------
extern "C" void kernel_launch(void* const* d_in, const int* in_sizes, int n_in,
                              void* d_out, int out_size, void* d_ws, size_t ws_size,
                              hipStream_t stream) {
  const float* X     = (const float*)d_in[0];
  const float* Wq    = (const float*)d_in[1];
  const float* bq    = (const float*)d_in[2];
  const float* Wv    = (const float*)d_in[3];
  const float* bv    = (const float*)d_in[4];
  const float* Wo    = (const float*)d_in[5];
  const float* bo    = (const float*)d_in[6];
  const float* Kemb  = (const float*)d_in[7];
  const float* Kkeys = (const float*)d_in[8];

  char*  w   = (char*)d_ws;
  size_t off = 0;
  auto alloc = [&](size_t bytes) {
    void* p = w + off;
    off += (bytes + 255) & ~(size_t)255;
    return p;
  };

  __bf16* Xb  = (__bf16*)alloc((size_t)BROWS * D_MODEL * 2);  // X bf16, reused as `retrieved`
  __bf16* Qb  = (__bf16*)alloc((size_t)BROWS * D_MODEL * 2);  // q bf16, reused as `value`
  __bf16* Wqb = (__bf16*)alloc((size_t)D_MODEL * D_MODEL * 2);
  __bf16* Wvb = (__bf16*)alloc((size_t)D_MODEL * D_MODEL * 2);
  __bf16* Wob = (__bf16*)alloc((size_t)NTOK * D_MODEL * D_MODEL * 2);

  auto cvt = [&](const float* s, __bf16* d, size_t n) {
    int n4 = (int)(n / 4);
    cvt_f32_bf16<<<(n4 + 255) / 256, 256, 0, stream>>>(s, d, n4);
  };
  cvt(X,  Xb,  (size_t)BROWS * D_MODEL);
  cvt(Wq, Wqb, (size_t)D_MODEL * D_MODEL);
  cvt(Wv, Wvb, (size_t)D_MODEL * D_MODEL);
  cvt(Wo, Wob, (size_t)NTOK * D_MODEL * D_MODEL);

  dim3 blk(256);
  // q = X @ Wq^T + bq
  gemm_nt_bf16<0><<<dim3(BROWS / 256, D_MODEL / 64), blk, 0, stream>>>(
      Xb, Wqb, bq, (void*)Qb, BROWS, D_MODEL, D_MODEL);
  // scores / top-3 / softmax / gather  -> retrieved (overwrites Xb)
  retrieve_topk<<<BROWS / 8, blk, 0, stream>>>(Qb, Kkeys, Kemb, Xb);
  // value = retrieved @ Wv^T + bv  -> overwrites Qb
  gemm_nt_bf16<0><<<dim3(BROWS / 256, D_MODEL / 64), blk, 0, stream>>>(
      Xb, Wvb, bv, (void*)Qb, BROWS, D_MODEL, D_MODEL);
  // tokens = value @ Wo^T + bo -> d_out, (2, B, d) permuted store
  gemm_nt_bf16<1><<<dim3(BROWS / 256, (NTOK * D_MODEL) / 64), blk, 0, stream>>>(
      Qb, Wob, bo, d_out, BROWS, NTOK * D_MODEL, D_MODEL);
}
